// MultiHeadAttention_29566554865892
// MI455X (gfx1250) — compile-verified, matrix operand order checked by reading, not run on hardware
//
#include <hip/hip_runtime.h>

#define DEV __device__ __forceinline__

typedef _Float16 v16h __attribute__((ext_vector_type(16)));
typedef _Float16 v8h  __attribute__((ext_vector_type(8)));
typedef _Float16 v4h  __attribute__((ext_vector_type(4)));
typedef float    v8f  __attribute__((ext_vector_type(8)));

static constexpr int BATCH = 2;
static constexpr int SEQ   = 2048;
static constexpr int DIM   = 1024;
static constexpr int HEADS = 16;
static constexpr int HDIM  = 64;

// D = A(16x32 f16) * B(32x16 f16) + C(16x16 f32)
DEV v8f wmma_f16(v16h a, v16h b, v8f c) {
  return __builtin_amdgcn_wmma_f32_16x16x32_f16(false, a, false, b, (short)0, c,
                                                false, false);
}

// A/B fragment loader (ISA 7.12.2, wave32):
//  lanes 0-15 : elems[0..7] = K 0..7,  elems[8..15] = K 16..23
//  lanes 16-31: elems[0..7] = K 8..15, elems[8..15] = K 24..31
// p points at (row, k-chunk base); data contiguous in K. Works for global or
// LDS-derived pointers (clang keeps the address space -> ds_load_b128 for LDS).
DEV v16h load_frag16(const _Float16* p, int halfid) {
  v8h lo = *(const v8h*)(p + 8 * halfid);
  v8h hi = *(const v8h*)(p + 16 + 8 * halfid);
  v16h r;
#pragma unroll
  for (int j = 0; j < 8; ++j) { r[j] = lo[j]; r[j + 8] = hi[j]; }
  return r;
}

DEV float redmax16(float v) {
  v = fmaxf(v, __shfl_xor(v, 1, 16));
  v = fmaxf(v, __shfl_xor(v, 2, 16));
  v = fmaxf(v, __shfl_xor(v, 4, 16));
  v = fmaxf(v, __shfl_xor(v, 8, 16));
  return v;
}
DEV float redsum16(float v) {
  v += __shfl_xor(v, 1, 16);
  v += __shfl_xor(v, 2, 16);
  v += __shfl_xor(v, 4, 16);
  v += __shfl_xor(v, 8, 16);
  return v;
}

// ---------------------------------------------------------------------------
// Kernel 1: W[k][n] (f32, row-major) -> WT[n][k] (f16) for the 4 weights.
// LDS-tiled 32x32 transpose; grid (32, 32, 4), block (32, 8).
// ---------------------------------------------------------------------------
__global__ void transpose_weights_kernel(const float* __restrict__ Wq,
                                         const float* __restrict__ Wk,
                                         const float* __restrict__ Wv,
                                         const float* __restrict__ Wo,
                                         _Float16* __restrict__ WqT,
                                         _Float16* __restrict__ WkT,
                                         _Float16* __restrict__ WvT,
                                         _Float16* __restrict__ WoT) {
  __shared__ float tile[32][33];
  const int z = blockIdx.z;
  const float* W = (z == 0) ? Wq : (z == 1) ? Wk : (z == 2) ? Wv : Wo;
  _Float16* WT   = (z == 0) ? WqT : (z == 1) ? WkT : (z == 2) ? WvT : WoT;

  const int tx = threadIdx.x, ty = threadIdx.y;
  const int n0 = blockIdx.x * 32;  // column block of W
  const int k0 = blockIdx.y * 32;  // row block of W
#pragma unroll
  for (int r = 0; r < 4; ++r)
    tile[ty + 8 * r][tx] = W[(size_t)(k0 + ty + 8 * r) * DIM + (n0 + tx)];
  __syncthreads();
#pragma unroll
  for (int r = 0; r < 4; ++r)   // WT[n][k] = W[k][n]; contiguous in tx (=k)
    WT[(size_t)(n0 + ty + 8 * r) * DIM + (k0 + tx)] =
        (_Float16)tile[tx][ty + 8 * r];
}

// ---------------------------------------------------------------------------
// Kernel 2: x f32 -> f16 (elementwise). grid 4096, block 256, 4 elems/thread.
// ---------------------------------------------------------------------------
__global__ void convert_x_kernel(const float* __restrict__ x,
                                 _Float16* __restrict__ xh) {
  const size_t i = ((size_t)blockIdx.x * blockDim.x + threadIdx.x) * 4;
  float4 v = *(const float4*)(x + i);
  v4h h;
  h[0] = (_Float16)v.x; h[1] = (_Float16)v.y;
  h[2] = (_Float16)v.z; h[3] = (_Float16)v.w;
  *(v4h*)(xh + i) = h;
}

// ---------------------------------------------------------------------------
// Kernel 3: QKV projections. M = B*S = 4096, N = K = 1024.
// grid (64, 16, 3): 64x64 output tile per block, z selects Q/K/V.
// block 128 = 4 waves; wave w owns rows [16w,16w+16) of the tile.
// Weight chunk (32k x 64n) staged in LDS once per block, double-buffered:
// all 4 waves share it (4x less L2 traffic, ds_load_b128 fragment reads).
// Q,K written [B,H,S,Dh] f16; V written transposed [B,H,Dh,S] f16.
// ---------------------------------------------------------------------------
__global__ void qkv_gemm_kernel(const _Float16* __restrict__ xh,
                                const _Float16* __restrict__ WqT,
                                const _Float16* __restrict__ WkT,
                                const _Float16* __restrict__ WvT,
                                const float* __restrict__ bq,
                                const float* __restrict__ bk,
                                const float* __restrict__ bv,
                                _Float16* __restrict__ Qo,
                                _Float16* __restrict__ Ko,
                                _Float16* __restrict__ VTo) {
  // 64 rows (n) x 32 halves (k), row stride 40 halves: 16-lane frag reads map
  // to banks (ln*20)%64 -> 16 distinct banks. 2 buffers = 10 KB LDS.
  __shared__ _Float16 bstage[2][64 * 40];

  const int z = blockIdx.z;
  const _Float16* WT  = (z == 0) ? WqT : (z == 1) ? WkT : WvT;
  const float*    bia = (z == 0) ? bq  : (z == 1) ? bk  : bv;

  const int wave = threadIdx.x >> 5;
  const int lane = threadIdx.x & 31;
  const int halfid = lane >> 4;
  const int ln = lane & 15;
  const int mbase = blockIdx.x * 64 + wave * 16;
  const int nbase = blockIdx.y * 64;
  const size_t arow = (size_t)(mbase + ln) * DIM;

  // cooperative stage of one 32k x 64n weight chunk: 128 threads x 32 bytes
  auto stage = [&](int buf, int kks) {
    const int t = threadIdx.x;
    const int n = t >> 1, seg = t & 1;
    const _Float16* src = WT + (size_t)(nbase + n) * DIM + kks + seg * 16;
    v8h x0 = *(const v8h*)(src);
    v8h x1 = *(const v8h*)(src + 8);
    *(v8h*)(&bstage[buf][n * 40 + seg * 16]) = x0;
    *(v8h*)(&bstage[buf][n * 40 + seg * 16 + 8]) = x1;
  };

  v8f acc[4];
#pragma unroll
  for (int t = 0; t < 4; ++t)
#pragma unroll
    for (int j = 0; j < 8; ++j) acc[t][j] = 0.0f;

  stage(0, 0);
  __syncthreads();
  for (int kk = 0; kk < DIM; kk += 32) {
    const int cur = (kk >> 5) & 1;
    if (kk + 32 < DIM) stage(cur ^ 1, kk + 32);   // prefetch next chunk
    v16h a = load_frag16(xh + arow + kk, halfid);
#pragma unroll
    for (int t = 0; t < 4; ++t) {
      v16h b = load_frag16(&bstage[cur][(t * 16 + ln) * 40], halfid);
      acc[t] = wmma_f16(a, b, acc[t]);
    }
    __syncthreads();
  }

#pragma unroll
  for (int t = 0; t < 4; ++t) {
    const int col = nbase + t * 16 + ln;     // this lane's output column
    const float bias = bia[col];
    const int h = col >> 6, dh = col & 63;
#pragma unroll
    for (int i = 0; i < 8; ++i) {
      const int m = mbase + i + 8 * halfid;  // global row (token index)
      const int b = m >> 11, s = m & (SEQ - 1);
      const float val = acc[t][i] + bias;
      if (z < 2) {  // Q, K : [B,H,S,Dh]
        _Float16* dst = (z == 0) ? Qo : Ko;
        dst[(((size_t)(b * HEADS + h)) * SEQ + s) * HDIM + dh] = (_Float16)val;
      } else {      // V transposed : [B,H,Dh,S]
        VTo[(((size_t)(b * HEADS + h)) * HDIM + dh) * SEQ + s] = (_Float16)val;
      }
    }
  }
}

// ---------------------------------------------------------------------------
// Kernel 4: flash attention. grid (S/32=64, H=16, B=2), block 32 (one wave).
// Each wave owns TWO 16-query tiles (32 queries) so every K/V fragment loaded
// from L2 feeds two WMMAs (2x arithmetic intensity). Per 32-key chunk:
//   8 WMMA for S = Q*K^T, online softmax (width-16 shfl row reductions),
//   P restaged via LDS to A-fragment layout, 8 WMMA for O += P*V.
// ---------------------------------------------------------------------------
__global__ void flash_attn_kernel(const _Float16* __restrict__ Q,
                                  const _Float16* __restrict__ K,
                                  const _Float16* __restrict__ VT,
                                  _Float16* __restrict__ attn_out /*[B*S, D]*/) {
  __shared__ _Float16 lp[2][16 * 40];  // two 16x32 P tiles, padded rows

  const int qt = blockIdx.x, h = blockIdx.y, b = blockIdx.z;
  const size_t bh = (size_t)(b * HEADS + h);
  const _Float16* Qbh = Q + bh * SEQ * HDIM;
  const _Float16* Kbh = K + bh * SEQ * HDIM;
  const _Float16* Vbh = VT + bh * HDIM * SEQ;

  const int lane = threadIdx.x;
  const int halfid = lane >> 4;
  const int ln = lane & 15;
  const float sm_scale = 0.125f;  // 1/sqrt(64)

  // Q A-fragments for both query tiles, full Dh=64, held in registers.
  v16h aq[2][2];
#pragma unroll
  for (int qi = 0; qi < 2; ++qi) {
    const size_t qrow = (size_t)(qt * 32 + qi * 16 + ln) * HDIM;
    aq[qi][0] = load_frag16(Qbh + qrow + 0, halfid);
    aq[qi][1] = load_frag16(Qbh + qrow + 32, halfid);
  }

  float m_i[2][8], l_i[2][8], alpha[2][8];
  v8f o[2][4];
#pragma unroll
  for (int qi = 0; qi < 2; ++qi) {
#pragma unroll
    for (int i = 0; i < 8; ++i) { m_i[qi][i] = -1e30f; l_i[qi][i] = 0.0f; }
#pragma unroll
    for (int t = 0; t < 4; ++t)
#pragma unroll
      for (int j = 0; j < 8; ++j) o[qi][t][j] = 0.0f;
  }

  for (int kb = 0; kb < SEQ; kb += 32) {
    if (kb + 32 < SEQ) {  // prefetch next chunk's K rows / V columns into L2/L0
      __builtin_prefetch(Kbh + (size_t)(kb + 32 + ln) * HDIM, 0, 1);
      __builtin_prefetch(Kbh + (size_t)(kb + 48 + ln) * HDIM, 0, 1);
      __builtin_prefetch(Vbh + (size_t)(lane * 2) * SEQ + kb + 32, 0, 1);
    }

    // ---- K^T B-fragments, shared by both query tiles ----
    const size_t kr0 = (size_t)(kb + ln) * HDIM;
    const size_t kr1 = (size_t)(kb + 16 + ln) * HDIM;
    v16h b0a = load_frag16(Kbh + kr0 + 0, halfid);
    v16h b0b = load_frag16(Kbh + kr0 + 32, halfid);
    v16h b1a = load_frag16(Kbh + kr1 + 0, halfid);
    v16h b1b = load_frag16(Kbh + kr1 + 32, halfid);

#pragma unroll
    for (int qi = 0; qi < 2; ++qi) {
      // ---- scores: two 16x16 tiles (keys kb..kb+15, kb+16..kb+31) ----
      v8f s0, s1;
#pragma unroll
      for (int j = 0; j < 8; ++j) { s0[j] = 0.0f; s1[j] = 0.0f; }
      s0 = wmma_f16(aq[qi][0], b0a, s0);
      s0 = wmma_f16(aq[qi][1], b0b, s0);
      s1 = wmma_f16(aq[qi][0], b1a, s1);
      s1 = wmma_f16(aq[qi][1], b1b, s1);

      // ---- online softmax over the 32 new keys ----
      float mnew[8];
#pragma unroll
      for (int i = 0; i < 8; ++i) {
        const float a0 = s0[i] * sm_scale, a1 = s1[i] * sm_scale;
        s0[i] = a0; s1[i] = a1;
        const float mx = redmax16(fmaxf(a0, a1));   // row = i + 8*halfid
        mnew[i]         = fmaxf(m_i[qi][i], mx);
        alpha[qi][i]    = __expf(m_i[qi][i] - mnew[i]);
        m_i[qi][i]      = mnew[i];
      }
#pragma unroll
      for (int i = 0; i < 8; ++i) {
        const float p0 = __expf(s0[i] - mnew[i]);
        const float p1 = __expf(s1[i] - mnew[i]);
        const int row = i + 8 * halfid;
        lp[qi][row * 40 + ln]      = (_Float16)p0;  // P[row][ln]
        lp[qi][row * 40 + 16 + ln] = (_Float16)p1;  // P[row][16+ln]
        l_i[qi][i] = l_i[qi][i] * alpha[qi][i] + redsum16(p0 + p1);
      }
    }
    __syncthreads();  // single wave -> S_NOP; orders LDS store->load

    // reload P tiles as 16x32 A-fragments: lane = row, contiguous in key dim
    v16h ap[2];
#pragma unroll
    for (int qi = 0; qi < 2; ++qi)
      ap[qi] = load_frag16(&lp[qi][ln * 40], halfid);
    __syncthreads();

    // ---- O = O*alpha + P @ V ; V fragments shared by both query tiles ----
#pragma unroll
    for (int t = 0; t < 4; ++t) {
      // B = V chunk (32 keys x 16 dh): VT row = dh, contiguous in key
      v16h bv = load_frag16(Vbh + (size_t)(t * 16 + ln) * SEQ + kb, halfid);
#pragma unroll
      for (int qi = 0; qi < 2; ++qi) {
#pragma unroll
        for (int i = 0; i < 8; ++i) o[qi][t][i] *= alpha[qi][i];
        o[qi][t] = wmma_f16(ap[qi], bv, o[qi][t]);
      }
    }
  }

  // ---- normalize and store f16 [B*S, D] for the output projection ----
#pragma unroll
  for (int qi = 0; qi < 2; ++qi)
#pragma unroll
    for (int t = 0; t < 4; ++t) {
      const int col = h * HDIM + t * 16 + ln;
#pragma unroll
      for (int i = 0; i < 8; ++i) {
        const size_t row = (size_t)b * SEQ + qt * 32 + qi * 16 + i + 8 * halfid;
        attn_out[row * DIM + col] = (_Float16)(o[qi][t][i] * (1.0f / l_i[qi][i]));
      }
    }
}

// ---------------------------------------------------------------------------
// Kernel 5: output projection, f32 result + bias. grid (64, 16), block 128.
// Same LDS double-buffered weight staging as the QKV GEMM.
// ---------------------------------------------------------------------------
__global__ void out_proj_kernel(const _Float16* __restrict__ attn,
                                const _Float16* __restrict__ WoT,
                                const float* __restrict__ bo,
                                float* __restrict__ out) {
  __shared__ _Float16 bstage[2][64 * 40];

  const int wave = threadIdx.x >> 5;
  const int lane = threadIdx.x & 31;
  const int halfid = lane >> 4;
  const int ln = lane & 15;
  const int mbase = blockIdx.x * 64 + wave * 16;
  const int nbase = blockIdx.y * 64;
  const size_t arow = (size_t)(mbase + ln) * DIM;

  auto stage = [&](int buf, int kks) {
    const int t = threadIdx.x;
    const int n = t >> 1, seg = t & 1;
    const _Float16* src = WoT + (size_t)(nbase + n) * DIM + kks + seg * 16;
    v8h x0 = *(const v8h*)(src);
    v8h x1 = *(const v8h*)(src + 8);
    *(v8h*)(&bstage[buf][n * 40 + seg * 16]) = x0;
    *(v8h*)(&bstage[buf][n * 40 + seg * 16 + 8]) = x1;
  };

  v8f acc[4];
#pragma unroll
  for (int t = 0; t < 4; ++t)
#pragma unroll
    for (int j = 0; j < 8; ++j) acc[t][j] = 0.0f;

  stage(0, 0);
  __syncthreads();
  for (int kk = 0; kk < DIM; kk += 32) {
    const int cur = (kk >> 5) & 1;
    if (kk + 32 < DIM) stage(cur ^ 1, kk + 32);
    v16h a = load_frag16(attn + arow + kk, halfid);
#pragma unroll
    for (int t = 0; t < 4; ++t) {
      v16h bfrag = load_frag16(&bstage[cur][(t * 16 + ln) * 40], halfid);
      acc[t] = wmma_f16(a, bfrag, acc[t]);
    }
    __syncthreads();
  }

#pragma unroll
  for (int t = 0; t < 4; ++t) {
    const int col = nbase + t * 16 + ln;
    const float bias = bo[col];
#pragma unroll
    for (int i = 0; i < 8; ++i) {
      const size_t m = (size_t)mbase + i + 8 * halfid;
      out[m * DIM + col] = acc[t][i] + bias;
    }
  }
}

// ---------------------------------------------------------------------------
// Host launcher
// ---------------------------------------------------------------------------
extern "C" void kernel_launch(void* const* d_in, const int* in_sizes, int n_in,
                              void* d_out, int out_size, void* d_ws, size_t ws_size,
                              hipStream_t stream) {
  const float* x  = (const float*)d_in[0];
  const float* Wq = (const float*)d_in[1];
  const float* bq = (const float*)d_in[2];
  const float* Wk = (const float*)d_in[3];
  const float* bk = (const float*)d_in[4];
  const float* Wv = (const float*)d_in[5];
  const float* bv = (const float*)d_in[6];
  const float* Wo = (const float*)d_in[7];
  const float* bo = (const float*)d_in[8];
  float* out = (float*)d_out;

  // Workspace layout (f16 elements). WU = 1M elems per weight, xh/Q/K/VT/attn
  // each 4M elems. Total 24M halves = 48 MB.
  _Float16* ws = (_Float16*)d_ws;
  const size_t WU = (size_t)DIM * DIM;           // 1M
  _Float16* WqT  = ws + 0 * WU;
  _Float16* WkT  = ws + 1 * WU;
  _Float16* WvT  = ws + 2 * WU;
  _Float16* WoT  = ws + 3 * WU;
  _Float16* xh   = ws + 4 * WU;                  // [B*S, D]
  _Float16* Qb   = ws + 8 * WU;                  // [B,H,S,Dh]
  _Float16* Kb   = ws + 12 * WU;                 // [B,H,S,Dh]
  _Float16* VTb  = ws + 16 * WU;                 // [B,H,Dh,S]
  _Float16* attn = ws + 20 * WU;                 // [B*S, D]

  transpose_weights_kernel<<<dim3(DIM / 32, DIM / 32, 4), dim3(32, 8), 0, stream>>>(
      Wq, Wk, Wv, Wo, WqT, WkT, WvT, WoT);

  convert_x_kernel<<<dim3((BATCH * SEQ * DIM) / 1024), dim3(256), 0, stream>>>(x, xh);

  qkv_gemm_kernel<<<dim3((BATCH * SEQ) / 64, DIM / 64, 3), dim3(128), 0, stream>>>(
      xh, WqT, WkT, WvT, bq, bk, bv, Qb, Kb, VTb);

  flash_attn_kernel<<<dim3(SEQ / 32, HEADS, BATCH), dim3(32), 0, stream>>>(
      Qb, Kb, VTb, attn);

  out_proj_kernel<<<dim3((BATCH * SEQ) / 64, DIM / 64), dim3(128), 0, stream>>>(
      attn, WoT, bo, out);
}
